// BiLSTM_NER_62534723830107
// MI455X (gfx1250) — compile-verified, matrix operand order checked by reading, not run on hardware
//
#include <hip/hip_runtime.h>
#include <stdint.h>

#define BB 64
#define TT 512
#define EE 256
#define HH 512
#define GG 2048            // 4*HH
#define KK 768             // EE + HH  (fused [x ; h] K dimension)
#define CC 17
#define NWG 16             // workgroups cooperating per (batch-tile, direction)

typedef __attribute__((ext_vector_type(16))) _Float16 v16h;
typedef __attribute__((ext_vector_type(8)))  float    v8f;

// ---- WMMA helpers -----------------------------------------------------------

// Load one 16-lane-half slice of a 16x32 (f16) A or 32x16 B tile, following the
// CDNA5 16-bit VGPR layout: lanes 0-15 carry K {0..7,16..23} of row (lane&15),
// lanes 16-31 carry K {8..15,24..31}.  Caller passes row pointer already offset
// by k0 + (lane>=16 ? 8 : 0); we fetch halves [0..7] and [16..23] from there.
__device__ __forceinline__ v16h ld_tile(const _Float16* p) {
    union { uint4 u[2]; v16h h; } t;
    t.u[0] = *(const uint4*)(p);
    t.u[1] = *(const uint4*)(p + 16);
    return t.h;
}

__device__ __forceinline__ v8f wmma16(v16h a, v16h b, v8f c) {
    // D = A(16x32 f16) * B(32x16 f16) + C(16x16 f32)
    return __builtin_amdgcn_wmma_f32_16x16x32_f16(
        /*neg_a=*/false, a, /*neg_b=*/false, b,
        /*c_mod=*/(short)0, c, /*reuse_a=*/false, /*reuse_b=*/false);
}

__device__ __forceinline__ float sigmoidf_(float v) {
    return 1.0f / (1.0f + __expf(-v));
}

// ---- Prep kernels (fp32 -> f16 weight repack) -------------------------------

__global__ void prep_wcat(const float* __restrict__ wihf, const float* __restrict__ whhf,
                          const float* __restrict__ wihb, const float* __restrict__ whhb,
                          _Float16* __restrict__ wcat) {
    int i = blockIdx.x * blockDim.x + threadIdx.x;
    if (i >= 2 * GG * KK) return;
    int dir = i / (GG * KK);
    int r   = i % (GG * KK);
    int g   = r / KK;
    int k   = r % KK;
    const float* wih = dir ? wihb : wihf;
    const float* whh = dir ? whhb : whhf;
    float v = (k < EE) ? wih[g * EE + k] : whh[g * HH + (k - EE)];
    wcat[i] = (_Float16)v;
}

__global__ void prep_bias(const float* __restrict__ bihf, const float* __restrict__ bhhf,
                          const float* __restrict__ bihb, const float* __restrict__ bhhb,
                          float* __restrict__ bias) {
    int i = blockIdx.x * blockDim.x + threadIdx.x;
    if (i >= 2 * GG) return;
    int dir = i / GG, g = i % GG;
    bias[i] = dir ? (bihb[g] + bhhb[g]) : (bihf[g] + bhhf[g]);
}

// fc_w (17 x 1024) zero-padded to 32 rows, f16
__global__ void prep_fc(const float* __restrict__ fcw, _Float16* __restrict__ fcp) {
    int i = blockIdx.x * blockDim.x + threadIdx.x;
    if (i >= 32 * 2 * HH) return;
    int n = i >> 10, k = i & (2 * HH - 1);
    fcp[i] = (_Float16)((n < CC) ? fcw[n * (2 * HH) + k] : 0.0f);
}

// ---- Embedding gather (fp32 table -> f16 activations) -----------------------

__global__ void embed_gather(const int* __restrict__ x, const float* __restrict__ emb,
                             _Float16* __restrict__ xe) {
    int i = (blockIdx.x * blockDim.x + threadIdx.x) * 4;   // < BB*TT*EE = 8388608
    int bt = i >> 8;
    int e  = i & (EE - 1);
    int tok = x[bt];
    float4 v = *(const float4*)(emb + (size_t)tok * EE + e);
    union { _Float16 h[4]; uint2 u; } o;
    o.h[0] = (_Float16)v.x; o.h[1] = (_Float16)v.y;
    o.h[2] = (_Float16)v.z; o.h[3] = (_Float16)v.w;
    *(uint2*)(xe + i) = o.u;
}

// ---- Cooperative BiLSTM kernel ---------------------------------------------
// Grid = 128 blocks of 256 threads: 8 groups ((4 batch tiles) x (2 dirs)), each
// group = 16 cooperating WGs.  A WG owns 32 hidden columns x 4 gates = 128
// weight rows (192 KB f16) staged ONCE into LDS; the per-step GEMM then reads
// only LDS (A: sh_x/sh_h, B: sh_w) -> no steady-state global weight traffic.
// h is exchanged through hcat[t] itself (distinct storage per step => no WAR),
// with one monotonic atomic barrier per group per step.

__global__ __launch_bounds__(256) void lstm_coop(
    const _Float16* __restrict__ xe,    // (B, T, E) f16
    const _Float16* __restrict__ wcat,  // (2, G, K=768) f16
    const float*    __restrict__ bias,  // (2, G) f32  (b_ih + b_hh)
    _Float16*       __restrict__ hcat,  // (B, T, 2H) f16 output / h exchange
    unsigned int*   __restrict__ sync)  // 8 zero-initialized group counters
{
    const int cid   = (int)blockIdx.x & (NWG - 1);   // WG id in group: 0..15
    const int group = (int)blockIdx.x >> 4;          // 0..7
    const int dir   = group & 1;
    const int btile = group >> 1;
    const int hid0  = cid * 32;                      // hidden slice [hid0,hid0+32)

    __shared__ __align__(16) _Float16 sh_w[8 * 16 * KK];   // 192 KB weights
    __shared__ __align__(16) _Float16 sh_h[16 * HH];       // 16 KB  full h(t-1)
    __shared__ __align__(16) _Float16 sh_x[16 * EE];       // 8 KB   x_t tile
    __shared__ __align__(16) float    sh_g[4 * 16 * 32];   // 8 KB   gate exchange

    const int tid  = threadIdx.x;
    const int lane = tid & 31;
    const int wv   = tid >> 5;            // 0..7
    const int q    = wv >> 1;             // gate 0..3 (i,f,g,o)
    const int u    = wv & 1;              // sub-tile (hidden cols u*16..u*16+15)
    const int l16  = lane & 15;
    const int hi   = lane >> 4;
    const int kh   = hi ? 8 : 0;

    const _Float16* wd = wcat + (size_t)dir * GG * KK;
    const float*    bd = bias + dir * GG;

    // ---- stage this WG's 128 weight rows into LDS (once) ----
    for (int i4 = tid; i4 < (8 * 16 * KK) / 8; i4 += 256) {
        int L    = i4 * 8;                 // half index into sh_w
        int tile = L / (16 * KK);          // 0..7  (= q*2+u)
        int rem  = L % (16 * KK);
        int r    = rem / KK;
        int k    = rem % KK;
        int grow = (tile >> 1) * HH + hid0 + (tile & 1) * 16 + r;
        *(uint4*)(sh_w + L) = *(const uint4*)(wd + (size_t)grow * KK + k);
    }

    const float bval = bd[q * HH + hid0 + u * 16 + l16];

    // persistent cell state: element e = tid + k*256 -> (m = e>>5, col = e&31)
    float creg0 = 0.0f, creg1 = 0.0f;

    for (int s = 0; s < TT; ++s) {
        const int t = dir ? (TT - 1 - s) : s;

        // ---- stage x_t (16x256) and full h(t_prev) (16x512) into LDS ----
        {
            #pragma unroll
            for (int v = 0; v < 2; ++v) {               // x tile: 16 halves/thread
                int L = tid * 16 + v * 8;
                int m = L >> 8, e = L & (EE - 1);
                *(uint4*)(sh_x + L) =
                    *(const uint4*)(xe + ((size_t)(btile * 16 + m) * TT + t) * EE + e);
            }
            if (s == 0) {
                for (int i = tid; i < 16 * HH / 8; i += 256)
                    *(uint4*)(sh_h + i * 8) = make_uint4(0, 0, 0, 0);
            } else {
                const int tp = dir ? (t + 1) : (t - 1);
                #pragma unroll
                for (int v = 0; v < 4; ++v) {           // h: 32 halves/thread
                    int L = tid * 32 + v * 8;
                    int m = L >> 9, col = L & (HH - 1);
                    *(uint4*)(sh_h + L) =
                        *(const uint4*)(hcat + ((size_t)(btile * 16 + m) * TT + tp) * (2 * HH)
                                        + (size_t)dir * HH + col);
                }
            }
        }
        __syncthreads();

        // ---- gate tile GEMM, all operands in LDS ----
        int ko = 0;
        asm volatile("" : "+v"(ko));   // keep sh_w reads loop-variant (no LICM/spill)
        const _Float16* srow   = sh_w + (wv * 16 + l16) * KK + ko;
        const _Float16* arow_x = sh_x + l16 * EE;
        const _Float16* arow_h = sh_h + l16 * HH;

        v8f a0 = {bval, bval, bval, bval, bval, bval, bval, bval};
        v8f a1 = {0.f, 0.f, 0.f, 0.f, 0.f, 0.f, 0.f, 0.f};
        #pragma unroll
        for (int k0 = 0; k0 < EE; k0 += 64) {
            a0 = wmma16(ld_tile(arow_x + k0 + kh),      ld_tile(srow + k0 + kh),      a0);
            a1 = wmma16(ld_tile(arow_x + k0 + 32 + kh), ld_tile(srow + k0 + 32 + kh), a1);
        }
        #pragma unroll
        for (int k0 = 0; k0 < HH; k0 += 64) {
            a0 = wmma16(ld_tile(arow_h + k0 + kh),      ld_tile(srow + EE + k0 + kh),      a0);
            a1 = wmma16(ld_tile(arow_h + k0 + 32 + kh), ld_tile(srow + EE + k0 + 32 + kh), a1);
        }

        // store gate tile (combining partials); rows m=j+8*hi, col u*16+l16
        #pragma unroll
        for (int j = 0; j < 8; ++j)
            sh_g[(q * 16 + j + 8 * hi) * 32 + u * 16 + l16] = a0[j] + a1[j];
        __syncthreads();

        // ---- cell update: 512 elements, 2 per thread; c stays in registers ----
        {
            int e0 = tid, m0 = e0 >> 5, c0 = e0 & 31;
            int e1 = tid + 256, m1 = e1 >> 5, c1 = e1 & 31;
            float i0 = sh_g[(0 * 16 + m0) * 32 + c0], f0 = sh_g[(1 * 16 + m0) * 32 + c0];
            float g0 = sh_g[(2 * 16 + m0) * 32 + c0], o0 = sh_g[(3 * 16 + m0) * 32 + c0];
            float i1 = sh_g[(0 * 16 + m1) * 32 + c1], f1 = sh_g[(1 * 16 + m1) * 32 + c1];
            float g1 = sh_g[(2 * 16 + m1) * 32 + c1], o1 = sh_g[(3 * 16 + m1) * 32 + c1];
            creg0 = sigmoidf_(f0) * creg0 + sigmoidf_(i0) * tanhf(g0);
            creg1 = sigmoidf_(f1) * creg1 + sigmoidf_(i1) * tanhf(g1);
            float h0 = sigmoidf_(o0) * tanhf(creg0);
            float h1 = sigmoidf_(o1) * tanhf(creg1);
            hcat[((size_t)(btile * 16 + m0) * TT + t) * (2 * HH) + (size_t)dir * HH + hid0 + c0] = (_Float16)h0;
            hcat[((size_t)(btile * 16 + m1) * TT + t) * (2 * HH) + (size_t)dir * HH + hid0 + c1] = (_Float16)h1;
        }

        // ---- group barrier: all 16 WGs finished writing their h(t) slices ----
        __syncthreads();
        if (tid == 0) {
            unsigned int* cnt = sync + group;
            __hip_atomic_fetch_add(cnt, 1u, __ATOMIC_RELEASE, __HIP_MEMORY_SCOPE_AGENT);
            const unsigned int target = (unsigned int)(NWG * (s + 1));
            while (__hip_atomic_load(cnt, __ATOMIC_ACQUIRE, __HIP_MEMORY_SCOPE_AGENT) < target)
                __builtin_amdgcn_s_sleep(2);
        }
        __syncthreads();
    }
}

// ---- FC: emissions = hcat @ fc_w^T + fc_b  (WMMA, C padded 17 -> 32) --------

__global__ __launch_bounds__(256) void fc_kernel(
    const _Float16* __restrict__ hcat, const _Float16* __restrict__ fcp,
    const float* __restrict__ fcb, float* __restrict__ emis)
{
    const int lane = threadIdx.x & 31;
    const int wv   = threadIdx.x >> 5;
    const int mtile = (int)blockIdx.x * 8 + wv;     // 0..2047 row tiles
    const int l16 = lane & 15, hi = lane >> 4, kh = hi ? 8 : 0;
    const int KH = 2 * HH;

    const _Float16* arow = hcat + ((size_t)mtile * 16 + l16) * KH;
    const _Float16* b0   = fcp + (size_t)(l16) * KH;        // cols 0..15
    const _Float16* b1   = fcp + (size_t)(16 + l16) * KH;   // cols 16..31

    v8f a0 = {0.f, 0.f, 0.f, 0.f, 0.f, 0.f, 0.f, 0.f};
    v8f a1 = {0.f, 0.f, 0.f, 0.f, 0.f, 0.f, 0.f, 0.f};
    #pragma unroll 4
    for (int k0 = 0; k0 < KH; k0 += 32) {
        v16h a = ld_tile(arow + k0 + kh);
        a0 = wmma16(a, ld_tile(b0 + k0 + kh), a0);
        a1 = wmma16(a, ld_tile(b1 + k0 + kh), a1);
    }
    float bb0 = fcb[l16];
    float bb1 = (16 + l16 < CC) ? fcb[16 + l16] : 0.0f;
    #pragma unroll
    for (int j = 0; j < 8; ++j) {
        int m = j + 8 * hi;
        size_t r = (size_t)mtile * 16 + m;          // = b*T + t
        emis[r * 32 + l16]      = a0[j] + bb0;
        emis[r * 32 + 16 + l16] = a1[j] + bb1;
    }
}

// ---- CRF NLL: one wave per batch row ----------------------------------------

__global__ __launch_bounds__(32) void crf_kernel(
    const int* __restrict__ tags, const unsigned char* __restrict__ mask,
    const float* __restrict__ emis, const float* __restrict__ start_trans,
    const float* __restrict__ trans, const float* __restrict__ end_trans,
    float* __restrict__ out)
{
    const int b = blockIdx.x;
    const int j = threadIdx.x;
    __shared__ float al[32], aln[32], str[CC * CC];

    for (int i = j; i < CC * CC; i += 32) str[i] = trans[i];
    const float* em = emis + (size_t)b * TT * 32;
    const unsigned char* mk = mask + (size_t)b * TT;
    if (j < CC) al[j] = start_trans[j] + em[j];
    __syncthreads();

    for (int t = 1; t < TT; ++t) {
        if (j < CC) {
            float mx = -3.4e38f;
            for (int i = 0; i < CC; ++i) mx = fmaxf(mx, al[i] + str[i * CC + j]);
            float sum = 0.0f;
            for (int i = 0; i < CC; ++i) sum += __expf(al[i] + str[i * CC + j] - mx);
            float nxt = mx + __logf(sum) + em[t * 32 + j];
            aln[j] = mk[t] ? nxt : al[j];
        }
        __syncthreads();
        if (j < CC) al[j] = aln[j];
        __syncthreads();
    }

    if (j < CC) aln[j] = al[j] + end_trans[j];
    __syncthreads();
    if (j == 0) {
        float mx = -3.4e38f;
        for (int i = 0; i < CC; ++i) mx = fmaxf(mx, aln[i]);
        float sum = 0.0f;
        for (int i = 0; i < CC; ++i) sum += __expf(aln[i] - mx);
        float logZ = mx + __logf(sum);

        const int* tg = tags + (size_t)b * TT;
        float score = start_trans[tg[0]] + em[tg[0]];
        for (int t = 1; t < TT; ++t)
            if (mk[t]) score += str[tg[t - 1] * CC + tg[t]] + em[t * 32 + tg[t]];
        int cnt = 0;
        for (int t = 0; t < TT; ++t) cnt += mk[t] ? 1 : 0;
        int li = cnt > 0 ? cnt - 1 : 0;
        score += end_trans[tg[li]];
        atomicAdd(out, logZ - score);
    }
}

// ---- Host side --------------------------------------------------------------

extern "C" void kernel_launch(void* const* d_in, const int* in_sizes, int n_in,
                              void* d_out, int out_size, void* d_ws, size_t ws_size,
                              hipStream_t stream) {
    (void)in_sizes; (void)n_in; (void)ws_size;
    const int*   x    = (const int*)d_in[0];
    const int*   tags = (const int*)d_in[1];
    const unsigned char* mask = (const unsigned char*)d_in[2];
    const float* emb  = (const float*)d_in[3];
    const float* wihf = (const float*)d_in[4];
    const float* whhf = (const float*)d_in[5];
    const float* bihf = (const float*)d_in[6];
    const float* bhhf = (const float*)d_in[7];
    const float* wihb = (const float*)d_in[8];
    const float* whhb = (const float*)d_in[9];
    const float* bihb = (const float*)d_in[10];
    const float* bhhb = (const float*)d_in[11];
    const float* fcw  = (const float*)d_in[12];
    const float* fcb  = (const float*)d_in[13];
    const float* st   = (const float*)d_in[14];
    const float* tr   = (const float*)d_in[15];
    const float* et   = (const float*)d_in[16];

    char* ws = (char*)d_ws;
    size_t off = 0;
    auto carve = [&](size_t bytes) -> void* {
        void* p = ws + off;
        off += (bytes + 255) & ~(size_t)255;
        return p;
    };
    _Float16* xe   = (_Float16*)carve((size_t)BB * TT * EE * 2);       // 16 MB
    _Float16* wcat = (_Float16*)carve((size_t)2 * GG * KK * 2);        // 6 MB
    float*    bias = (float*)   carve((size_t)2 * GG * 4);             // 16 KB
    _Float16* fcp  = (_Float16*)carve((size_t)32 * 2 * HH * 2);        // 64 KB
    _Float16* hcat = (_Float16*)carve((size_t)BB * TT * 2 * HH * 2);   // 64 MB
    float*    emis = (float*)   carve((size_t)BB * TT * 32 * 4);       // 4 MB
    unsigned int* sync = (unsigned int*)carve(8 * sizeof(unsigned int));

    hipMemsetAsync(d_out, 0, sizeof(float) * (size_t)out_size, stream);
    hipMemsetAsync(sync, 0, 8 * sizeof(unsigned int), stream);

    prep_wcat<<<(2 * GG * KK) / 256, 256, 0, stream>>>(wihf, whhf, wihb, whhb, wcat);
    prep_bias<<<(2 * GG) / 256, 256, 0, stream>>>(bihf, bhhf, bihb, bhhb, bias);
    prep_fc<<<(32 * 2 * HH) / 256, 256, 0, stream>>>(fcw, fcp);
    embed_gather<<<(BB * TT * EE / 4) / 256, 256, 0, stream>>>(x, emb, xe);
    lstm_coop<<<8 * NWG, 256, 0, stream>>>(xe, wcat, bias, hcat, sync);
    fc_kernel<<<(BB * TT / 16) / 8, 256, 0, stream>>>(hcat, fcp, fcb, emis);
    crf_kernel<<<BB, 32, 0, stream>>>(tags, mask, emis, st, tr, et, (float*)d_out);
}
